// MultiModalFusion_69681549410398
// MI455X (gfx1250) — compile-verified, matrix operand order checked by reading, not run on hardware
//
#include <hip/hip_runtime.h>
#include <hip/hip_bf16.h>

typedef __bf16 bf16_t;
typedef __attribute__((ext_vector_type(16))) __bf16 v16bf;
typedef __attribute__((ext_vector_type(8)))  __bf16 v8bf;
typedef __attribute__((ext_vector_type(8)))  float  v8f;

#define MODE_BF16 0
#define MODE_GATE 1
#define MODE_F32  2

// ---------------------------------------------------------------------------
// WMMA GEMM:  C[B x N] = A[B x K] (bf16, lda) @ W[N x K]^T (bf16, row-major)
//             + bias[N] (fp32), with fused epilogue.
// Workgroup = 8 waves, tile 128(M) x 128(N). Wave = 2(M) x 4(N) 16x16 tiles.
// ---------------------------------------------------------------------------
template<int MODE>
__global__ __launch_bounds__(256)
void gemm_wmma(const bf16_t* __restrict__ A, int lda,
               const bf16_t* __restrict__ W, int K,
               const float* __restrict__ bias,
               const float* __restrict__ S, int ldsrc,   // gate source (fp32)
               bf16_t* __restrict__ Ob, float* __restrict__ Of,
               int ldo, int col_off)
{
    const int lane = threadIdx.x & 31;
    const int wave = threadIdx.x >> 5;
    const int wm   = wave >> 1;                 // 0..3
    const int wn   = wave & 1;                  // 0..1
    const long row0 = (long)blockIdx.x * 128 + wm * 32;
    const int  col0 = blockIdx.y * 128 + wn * 64;

    const int lrow = lane & 15;                 // row (A) / col (B) inside tile
    const int kgrp = (lane >> 4) << 3;          // 0 or 8 (K-chunk base per lane half)

    v8f zero = {};
    v8f acc[2][4];
    #pragma unroll
    for (int i = 0; i < 2; ++i)
        #pragma unroll
        for (int j = 0; j < 4; ++j) acc[i][j] = zero;

    const bf16_t* Abase[2];
    Abase[0] = A + (row0 + lrow) * (long)lda;
    Abase[1] = A + (row0 + 16 + lrow) * (long)lda;
    const bf16_t* Wbase[4];
    #pragma unroll
    for (int j = 0; j < 4; ++j)
        Wbase[j] = W + (long)(col0 + j * 16 + lrow) * (long)K;

    for (int k = 0; k < K; k += 32) {
        v16bf a[2], b[4];
        #pragma unroll
        for (int i = 0; i < 2; ++i) {
            const bf16_t* p = Abase[i] + k + kgrp;
            ((v8bf*)&a[i])[0] = *(const v8bf*)(p);
            ((v8bf*)&a[i])[1] = *(const v8bf*)(p + 16);
        }
        #pragma unroll
        for (int j = 0; j < 4; ++j) {
            const bf16_t* p = Wbase[j] + k + kgrp;
            ((v8bf*)&b[j])[0] = *(const v8bf*)(p);
            ((v8bf*)&b[j])[1] = *(const v8bf*)(p + 16);
        }
        if (k + 32 < K) {                       // CDNA5 global_prefetch_b8
            __builtin_prefetch(Abase[0] + k + 32 + kgrp, 0, 1);
            __builtin_prefetch(Wbase[0] + k + 32 + kgrp, 0, 1);
        }
        #pragma unroll
        for (int i = 0; i < 2; ++i)
            #pragma unroll
            for (int j = 0; j < 4; ++j)
                acc[i][j] = __builtin_amdgcn_wmma_f32_16x16x32_bf16(
                    false, a[i], false, b[j], (short)0, acc[i][j], false, false);
    }

    // Epilogue. D layout: VGPR v, lanes 0-15 -> (M=v, N=lane), lanes 16-31 -> (M=v+8).
    const int rsel = (lane >> 4) << 3;
    #pragma unroll
    for (int i = 0; i < 2; ++i) {
        #pragma unroll
        for (int j = 0; j < 4; ++j) {
            const int c  = col0 + j * 16 + (lane & 15);
            const float bs = bias[c];
            #pragma unroll
            for (int v = 0; v < 8; ++v) {
                const long r = row0 + i * 16 + rsel + v;
                float val = acc[i][j][v] + bs;
                if (MODE == MODE_BF16) {
                    Ob[r * (long)ldo + col_off + c] = (bf16_t)val;
                } else if (MODE == MODE_GATE) {
                    float gate = 1.f / (1.f + __expf(-val));
                    float sv   = S[r * (long)ldsrc + c];
                    Ob[r * (long)ldo + col_off + c] = (bf16_t)(sv * gate);
                } else {
                    Of[r * (long)ldo + c] = val;
                }
            }
        }
    }
}

// ---------------------------------------------------------------------------
// LayerNorm(1024) + affine + ReLU, in place on bf16 rows.
// ---------------------------------------------------------------------------
__global__ __launch_bounds__(256)
void k_layernorm_relu(bf16_t* __restrict__ h,
                      const float* __restrict__ lng, const float* __restrict__ lnb)
{
    const int N = 1024;
    __shared__ float sm[2][8];
    const long row = blockIdx.x;
    bf16_t* hr = h + row * (long)N;
    float x[4], sum = 0.f, sum2 = 0.f;
    #pragma unroll
    for (int i = 0; i < 4; ++i) {
        x[i] = (float)hr[threadIdx.x + i * 256];
        sum += x[i]; sum2 += x[i] * x[i];
    }
    #pragma unroll
    for (int off = 16; off; off >>= 1) {
        sum  += __shfl_xor(sum,  off, 32);
        sum2 += __shfl_xor(sum2, off, 32);
    }
    const int wv = threadIdx.x >> 5;
    if ((threadIdx.x & 31) == 0) { sm[0][wv] = sum; sm[1][wv] = sum2; }
    __syncthreads();
    if (wv == 0) {
        float a  = (threadIdx.x < 8) ? sm[0][threadIdx.x] : 0.f;
        float b2 = (threadIdx.x < 8) ? sm[1][threadIdx.x] : 0.f;
        #pragma unroll
        for (int off = 4; off; off >>= 1) {
            a  += __shfl_xor(a,  off, 32);
            b2 += __shfl_xor(b2, off, 32);
        }
        if (threadIdx.x == 0) { sm[0][0] = a; sm[1][0] = b2; }
    }
    __syncthreads();
    const float mean = sm[0][0] * (1.f / N);
    const float var  = sm[1][0] * (1.f / N) - mean * mean;
    const float rstd = rsqrtf(var + 1e-5f);
    #pragma unroll
    for (int i = 0; i < 4; ++i) {
        const int c = threadIdx.x + i * 256;
        float v = (x[i] - mean) * rstd * lng[c] + lnb[c];
        hr[c] = (bf16_t)(v > 0.f ? v : 0.f);
    }
}

// ---------------------------------------------------------------------------
// fp32 -> bf16 converters
// ---------------------------------------------------------------------------
__global__ void k_cvt(const float* __restrict__ src, bf16_t* __restrict__ dst, long n)
{
    long i = (long)blockIdx.x * blockDim.x + threadIdx.x;
    const long stride = (long)gridDim.x * blockDim.x;
    for (; i < n; i += stride) dst[i] = (bf16_t)src[i];
}

// B x 512 fp32 -> bf16 into a B x 1024 buffer (cols [0,512))
__global__ void k_cvt_in(const float* __restrict__ src, bf16_t* __restrict__ dst)
{
    const long i = (long)blockIdx.x * 256 + threadIdx.x;   // total B*512
    const long r = i >> 9;
    const int  c = (int)(i & 511);
    dst[(r << 10) + c] = (bf16_t)src[i];
}

// ---------------------------------------------------------------------------
extern "C" void kernel_launch(void* const* d_in, const int* in_sizes, int n_in,
                              void* d_out, int out_size, void* d_ws, size_t ws_size,
                              hipStream_t stream)
{
    (void)in_sizes; (void)n_in; (void)out_size; (void)ws_size;
    const long Bsz = 32768, H = 512;

    const float* g = (const float*)d_in[0];
    const float* s = (const float*)d_in[1];
    const float* t = (const float*)d_in[2];
    const float* g2s_w_in  = (const float*)d_in[3];
    const float* g2s_b_in  = (const float*)d_in[4];
    const float* g2s_w_out = (const float*)d_in[5];
    const float* g2s_b_out = (const float*)d_in[6];
    const float* s2g_w_in  = (const float*)d_in[7];
    const float* s2g_b_in  = (const float*)d_in[8];
    const float* s2g_w_out = (const float*)d_in[9];
    const float* s2g_b_out = (const float*)d_in[10];
    const float* t2m_w_in  = (const float*)d_in[11];
    const float* t2m_b_in  = (const float*)d_in[12];
    const float* t2m_w_out = (const float*)d_in[13];
    const float* t2m_b_out = (const float*)d_in[14];
    const float* gate_g_w  = (const float*)d_in[15];
    const float* gate_g_b  = (const float*)d_in[16];
    const float* gate_s_w  = (const float*)d_in[17];
    const float* gate_s_b  = (const float*)d_in[18];
    const float* gate_t_w  = (const float*)d_in[19];
    const float* gate_t_b  = (const float*)d_in[20];
    const float* fc1_w = (const float*)d_in[21];
    const float* fc1_b = (const float*)d_in[22];
    const float* ln_g  = (const float*)d_in[23];
    const float* ln_b  = (const float*)d_in[24];
    const float* fc2_w = (const float*)d_in[25];
    const float* fc2_b = (const float*)d_in[26];
    float* out = (float*)d_out;

    // workspace carve-up (all bf16)
    bf16_t* w = (bf16_t*)d_ws;
    long off = 0;
    auto alloc = [&](long n) { bf16_t* p = w + off; off += n; return p; };
    bf16_t* Wv_g2s = alloc(H * H);
    bf16_t* Wo_g2s = alloc(H * H);
    bf16_t* Wv_s2g = alloc(H * H);
    bf16_t* Wo_s2g = alloc(H * H);
    bf16_t* Wv_t2m = alloc(H * H);
    bf16_t* Wo_t2m = alloc(H * H);
    bf16_t* Gg = alloc(H * 2 * H);
    bf16_t* Gs = alloc(H * 2 * H);
    bf16_t* Gt = alloc(H * 2 * H);
    bf16_t* F1 = alloc(2 * H * 3 * H);
    bf16_t* F2 = alloc(H * 2 * H);
    bf16_t* cat_g = alloc(Bsz * 1024);   // [g | seq_to_graph]
    bf16_t* cat_s = alloc(Bsz * 1024);   // [s | graph_to_seq]
    bf16_t* cat_t = alloc(Bsz * 1024);   // [t | t2m]
    bf16_t* tmp   = alloc(Bsz * 512);    // v-projection scratch
    bf16_t* cat3  = alloc(Bsz * 1536);   // [gated_g | gated_s | gated_t]
    bf16_t* hraw  = alloc(Bsz * 1024);   // fc1 out -> LN in-place

    // --- weight conversions (fp32 -> bf16) ---
    auto cvt = [&](const float* src, bf16_t* dst, long n) {
        int blocks = (int)((n + 255) / 256);
        if (blocks > 32768) blocks = 32768;
        hipLaunchKernelGGL(k_cvt, dim3(blocks), dim3(256), 0, stream, src, dst, n);
    };
    cvt(g2s_w_in + 2 * H * H, Wv_g2s, H * H);
    cvt(g2s_w_out,            Wo_g2s, H * H);
    cvt(s2g_w_in + 2 * H * H, Wv_s2g, H * H);
    cvt(s2g_w_out,            Wo_s2g, H * H);
    cvt(t2m_w_in + 2 * H * H, Wv_t2m, H * H);
    cvt(t2m_w_out,            Wo_t2m, H * H);
    cvt(gate_g_w, Gg, H * 2 * H);
    cvt(gate_s_w, Gs, H * 2 * H);
    cvt(gate_t_w, Gt, H * 2 * H);
    cvt(fc1_w, F1, 2 * H * 3 * H);
    cvt(fc2_w, F2, H * 2 * H);

    // --- input conversions into concat buffers (cols [0,512)) ---
    const int cvtin_blocks = (int)(Bsz * 512 / 256);
    hipLaunchKernelGGL(k_cvt_in, dim3(cvtin_blocks), dim3(256), 0, stream, g, cat_g);
    hipLaunchKernelGGL(k_cvt_in, dim3(cvtin_blocks), dim3(256), 0, stream, s, cat_s);
    hipLaunchKernelGGL(k_cvt_in, dim3(cvtin_blocks), dim3(256), 0, stream, t, cat_t);

    const dim3 blk(256);
    const dim3 g512((unsigned)(Bsz / 128), 4);    // N = 512
    const dim3 g1024((unsigned)(Bsz / 128), 8);   // N = 1024

    // attention collapses: out = (v @ Wv^T + bv) @ Wout^T + bout
    hipLaunchKernelGGL((gemm_wmma<MODE_BF16>), g512, blk, 0, stream,
        cat_g, 1024, Wv_g2s, 512, g2s_b_in + 1024, (const float*)nullptr, 0,
        tmp, (float*)nullptr, 512, 0);
    hipLaunchKernelGGL((gemm_wmma<MODE_BF16>), g512, blk, 0, stream,
        tmp, 512, Wo_g2s, 512, g2s_b_out, (const float*)nullptr, 0,
        cat_s, (float*)nullptr, 1024, 512);                    // graph_to_seq
    hipLaunchKernelGGL((gemm_wmma<MODE_BF16>), g512, blk, 0, stream,
        cat_s, 1024, Wv_s2g, 512, s2g_b_in + 1024, (const float*)nullptr, 0,
        tmp, (float*)nullptr, 512, 0);
    hipLaunchKernelGGL((gemm_wmma<MODE_BF16>), g512, blk, 0, stream,
        tmp, 512, Wo_s2g, 512, s2g_b_out, (const float*)nullptr, 0,
        cat_g, (float*)nullptr, 1024, 512);                    // seq_to_graph
    hipLaunchKernelGGL((gemm_wmma<MODE_BF16>), g512, blk, 0, stream,
        cat_t, 1024, Wv_t2m, 512, t2m_b_in + 1024, (const float*)nullptr, 0,
        tmp, (float*)nullptr, 512, 0);
    hipLaunchKernelGGL((gemm_wmma<MODE_BF16>), g512, blk, 0, stream,
        tmp, 512, Wo_t2m, 512, t2m_b_out, (const float*)nullptr, 0,
        cat_t, (float*)nullptr, 1024, 512);                    // t2m (== mean)

    // gates: sigmoid(cat @ Gx^T + b) * src -> cat3 slots
    hipLaunchKernelGGL((gemm_wmma<MODE_GATE>), g512, blk, 0, stream,
        cat_g, 1024, Gg, 1024, gate_g_b, g, 512,
        cat3, (float*)nullptr, 1536, 0);
    hipLaunchKernelGGL((gemm_wmma<MODE_GATE>), g512, blk, 0, stream,
        cat_s, 1024, Gs, 1024, gate_s_b, s, 512,
        cat3, (float*)nullptr, 1536, 512);
    hipLaunchKernelGGL((gemm_wmma<MODE_GATE>), g512, blk, 0, stream,
        cat_t, 1024, Gt, 1024, gate_t_b, t, 512,
        cat3, (float*)nullptr, 1536, 1024);

    // fc1 -> LayerNorm+ReLU -> fc2
    hipLaunchKernelGGL((gemm_wmma<MODE_BF16>), g1024, blk, 0, stream,
        cat3, 1536, F1, 1536, fc1_b, (const float*)nullptr, 0,
        hraw, (float*)nullptr, 1024, 0);
    hipLaunchKernelGGL(k_layernorm_relu, dim3((unsigned)Bsz), blk, 0, stream,
        hraw, ln_g, ln_b);
    hipLaunchKernelGGL((gemm_wmma<MODE_F32>), g512, blk, 0, stream,
        hraw, 1024, F2, 1024, fc2_b, (const float*)nullptr, 0,
        (bf16_t*)nullptr, out, 512, 0);
}
